// GATIB_29102698398305
// MI455X (gfx1250) — matrix operand order repaired and checked by reading, not compiled, not run on hardware
//
#include <hip/hip_runtime.h>
#include <hip/hip_bf16.h>
#include <cstdint>

typedef __attribute__((ext_vector_type(4)))  _Float16 v4h;
typedef __attribute__((ext_vector_type(8)))  _Float16 v8h;
typedef __attribute__((ext_vector_type(16))) _Float16 v16h;
typedef __attribute__((ext_vector_type(4)))  float    v4f;
typedef __attribute__((ext_vector_type(8)))  float    v8f;

constexpr int   NN   = 20000;     // nodes
constexpr int   NE   = 640000;    // edges
constexpr int   DM   = 128;       // d_model (per-head channels)
constexpr int   NH   = 8;         // heads (layer 0)
constexpr int   NB   = 8;         // batch
constexpr int   BN   = NB * NN;   // 160000 flattened rows (divisible by 16)
constexpr float LEAK = 0.2f;
// encode(-inf) for the order-preserving float->uint mapping used by atomicMax
constexpr unsigned ENC_NEG_INF = 0x007FFFFFu;

__device__ __forceinline__ unsigned encodeOrd(float f) {
    unsigned u = __float_as_uint(f);
    return (u & 0x80000000u) ? ~u : (u | 0x80000000u);
}
__device__ __forceinline__ float decodeOrd(unsigned u) {
    return __uint_as_float((u & 0x80000000u) ? (u ^ 0x80000000u) : ~u);
}

// ---------------------------------------------------------------------------
// small utility kernels
// ---------------------------------------------------------------------------
__global__ void k_f32_to_f16(const float* __restrict__ in, _Float16* __restrict__ out, int n) {
    int i = blockIdx.x * blockDim.x + threadIdx.x;
    if (i < n) out[i] = (_Float16)in[i];
}

// Wt[n*K + k] = (f16) W[k*N + n]   (row-major [K,N] -> [N,K] f16)
__global__ void k_transpose_f16(const float* __restrict__ W, _Float16* __restrict__ Wt,
                                int K, int N) {
    int i = blockIdx.x * blockDim.x + threadIdx.x;
    if (i >= K * N) return;
    int n = i / K, k = i % K;
    Wt[(size_t)n * K + k] = (_Float16)W[(size_t)k * N + n];
}

__global__ void k_fill_u32(unsigned* __restrict__ p, unsigned v, int n) {
    int i = blockIdx.x * blockDim.x + threadIdx.x;
    if (i < n) p[i] = v;
}

// Y[i] = bias[i % rowLen]
__global__ void k_bias_bcast(float* __restrict__ Y, const float* __restrict__ bias,
                             int rowLen, int total) {
    int i = blockIdx.x * blockDim.x + threadIdx.x;
    if (i < total) Y[i] = bias[i % rowLen];
}

// x1 = elu(y0) stored as f16 (reuses the h0 buffer)
__global__ void k_elu_f16(const float* __restrict__ in, _Float16* __restrict__ out, int n) {
    int i = blockIdx.x * blockDim.x + threadIdx.x;
    if (i >= n) return;
    float v = in[i];
    out[i] = (_Float16)(v > 0.f ? v : expm1f(v));
}

// ---------------------------------------------------------------------------
// WMMA GEMM: C[M,Nout] = A[M,K] (f16, row-major) * B[K,Nout]
// B supplied pre-transposed f16: Bt[n*K + k].
// One wave computes a 16x64 tile: 4 accumulators, 4 wmma per 32-wide k-step.
// Fragment layouts used by this kernel (wave32, contiguous k order for A and B):
//   A 16x32 f16 : lane<16 -> M=lane,   halves[0..15]=K 0..15
//                 lane>=16 -> M=lane-16, halves[0..15]=K 16..31
//   B 32x16 f16 : lane<16 -> N=lane,   halves[0..15]=K 0..15
//                 lane>=16 -> N=lane-16, halves[0..15]=K 16..31
//   C/D 16x16 f32: vgpr v -> M = v + 8*(lane>=16), N = lane&15
// ---------------------------------------------------------------------------
__global__ void __launch_bounds__(128) k_gemm_wmma_f16(
    const _Float16* __restrict__ A, const _Float16* __restrict__ Bt,
    _Float16* __restrict__ outH, float* __restrict__ outF, int K, int Nout) {
    const int lane = threadIdx.x & 31;
    const int wave = threadIdx.x >> 5;
    const int wpb  = blockDim.x >> 5;
    const int m0   = blockIdx.x * 16;
    const int nb   = (blockIdx.y * wpb + wave) * 64;
    const int l15  = lane & 15;
    const int hi   = lane >> 4;               // 0 or 1

    const _Float16* aRow = A + (size_t)(m0 + l15) * K + hi * 16;
    const _Float16* bCol[4];
#pragma unroll
    for (int j = 0; j < 4; ++j)
        bCol[j] = Bt + (size_t)(nb + 16 * j + l15) * K + hi * 16;

    v8f acc[4] = {};
    const int kSteps = K >> 5;
    for (int kt = 0; kt < kSteps; ++kt) {
        if (kt + 1 < kSteps)                    // gfx1250 global_prefetch_b8
            __builtin_prefetch(aRow + (kt + 1) * 32, 0, 3);
        v8h alo = *(const v8h*)(aRow + kt * 32);
        v8h ahi = *(const v8h*)(aRow + kt * 32 + 8);
        v16h af;
#pragma unroll
        for (int i = 0; i < 8; ++i) { af[i] = alo[i]; af[i + 8] = ahi[i]; }
#pragma unroll
        for (int j = 0; j < 4; ++j) {
            v16h bf = *(const v16h*)(bCol[j] + kt * 32);
            acc[j] = __builtin_amdgcn_wmma_f32_16x16x32_f16(
                false, af, false, bf, (short)0, acc[j], false, false);
        }
    }
#pragma unroll
    for (int j = 0; j < 4; ++j) {
        const int c = nb + 16 * j + l15;
#pragma unroll
        for (int v = 0; v < 8; ++v) {
            const int r = m0 + v + hi * 8;
            if (outH) outH[(size_t)r * Nout + c] = (_Float16)acc[j][v];
            else      outF[(size_t)r * Nout + c] = acc[j][v];
        }
    }
}

// ---------------------------------------------------------------------------
// attention dots: one wave per (b*N+n, head):
//   eS = <h[row,head,:], a_src[head,:]>, eD likewise. 4 channels per lane.
// ---------------------------------------------------------------------------
template <typename T>
__global__ void k_edot(const T* __restrict__ Hm, const float* __restrict__ aS,
                       const float* __restrict__ aD, float* __restrict__ eS,
                       float* __restrict__ eD, int total /*BN*Hh*/, int Hh) {
    const int wid  = (blockIdx.x * blockDim.x + threadIdx.x) >> 5;
    const int lane = threadIdx.x & 31;
    if (wid >= total) return;
    const int h = wid % Hh;
    const size_t base = (size_t)(wid / Hh) * (size_t)Hh * DM + (size_t)h * DM + lane * 4;
    float s = 0.f, d = 0.f;
#pragma unroll
    for (int i = 0; i < 4; ++i) {
        const float hv = (float)Hm[base + i];
        s += hv * aS[h * DM + lane * 4 + i];
        d += hv * aD[h * DM + lane * 4 + i];
    }
#pragma unroll
    for (int o = 16; o; o >>= 1) { s += __shfl_xor(s, o); d += __shfl_xor(d, o); }
    if (lane == 0) { eS[wid] = s; eD[wid] = d; }
}

// ---------------------------------------------------------------------------
// segment-softmax pass 1: running max per (b,dst,h) via ordered-uint atomicMax
// ---------------------------------------------------------------------------
__global__ void k_edge_max(const int* __restrict__ src, const int* __restrict__ dst,
                           const float* __restrict__ eS, const float* __restrict__ eD,
                           unsigned* __restrict__ mEnc, int Hh, int total) {
    const int tid = blockIdx.x * blockDim.x + threadIdx.x;
    if (tid >= total) return;
    const int h = tid % Hh;
    const int e = (tid / Hh) % NE;
    const int b = tid / (Hh * NE);
    float v = eS[(b * NN + src[e]) * Hh + h] + eD[(b * NN + dst[e]) * Hh + h];
    v = v > 0.f ? v : LEAK * v;
    atomicMax(&mEnc[(b * NN + dst[e]) * Hh + h], encodeOrd(v));
}

// pass 2: numerator = exp(v - max); denom += numerator
__global__ void k_edge_exp(const int* __restrict__ src, const int* __restrict__ dst,
                           const float* __restrict__ eS, const float* __restrict__ eD,
                           const unsigned* __restrict__ mEnc, float* __restrict__ exBuf,
                           float* __restrict__ denom, int Hh, int total) {
    const int tid = blockIdx.x * blockDim.x + threadIdx.x;
    if (tid >= total) return;
    const int h = tid % Hh;
    const int e = (tid / Hh) % NE;
    const int b = tid / (Hh * NE);
    const int di = (b * NN + dst[e]) * Hh + h;
    float v = eS[(b * NN + src[e]) * Hh + h] + eD[di];
    v = v > 0.f ? v : LEAK * v;
    const float ex = __expf(v - decodeOrd(mEnc[di]));
    exBuf[tid] = ex;
    atomicAdd(&denom[di], ex);
}

// ---------------------------------------------------------------------------
// scatter-aggregate: 32 lanes per (b,e,h), 4 channels per lane.
//   Y[b,dst,h,:] += alpha * Hsrc[b,src,h,:]; lane0 adds alpha*scale to alphaOut
// ---------------------------------------------------------------------------
template <bool SRC_F16>
__global__ void k_agg(const int* __restrict__ src, const int* __restrict__ dst,
                      const float* __restrict__ exBuf, const float* __restrict__ denom,
                      const void* __restrict__ Hsrc, float* __restrict__ Y,
                      float* __restrict__ alphaOut, float alphaScale,
                      int Hh, int total /* B*NE*Hh*32 */) {
    const int tid = blockIdx.x * blockDim.x + threadIdx.x;
    if (tid >= total) return;
    const int lane = tid & 31;
    const int key  = tid >> 5;
    const int h = key % Hh;
    const int e = (key / Hh) % NE;
    const int b = key / (Hh * NE);
    const int s = src[e], d = dst[e];
    const float alpha = exBuf[key] / (denom[(b * NN + d) * Hh + h] + 1e-16f);
    const size_t so = ((size_t)(b * NN + s) * Hh + h) * DM + lane * 4;
    const size_t do_ = ((size_t)(b * NN + d) * Hh + h) * DM + lane * 4;
    float hv[4];
    if (SRC_F16) {
        v4h x = *(const v4h*)((const _Float16*)Hsrc + so);
#pragma unroll
        for (int i = 0; i < 4; ++i) hv[i] = (float)x[i];
    } else {
        v4f x = *(const v4f*)((const float*)Hsrc + so);
#pragma unroll
        for (int i = 0; i < 4; ++i) hv[i] = x[i];
    }
#pragma unroll
    for (int i = 0; i < 4; ++i) atomicAdd(Y + do_ + i, alpha * hv[i]);
    if (alphaOut && lane == 0) atomicAdd(alphaOut + e, alpha * alphaScale);
}

// ---------------------------------------------------------------------------
extern "C" void kernel_launch(void* const* d_in, const int* in_sizes, int n_in,
                              void* d_out, int out_size, void* d_ws, size_t ws_size,
                              hipStream_t stream) {
    const float* x   = (const float*)d_in[0];
    const int*   ei  = (const int*)d_in[1];
    const int*   src = ei;
    const int*   dst = ei + NE;
    const float* W0  = (const float*)d_in[2];
    const float* as0 = (const float*)d_in[3];
    const float* ad0 = (const float*)d_in[4];
    const float* b0  = (const float*)d_in[5];
    const float* W1  = (const float*)d_in[6];
    const float* as1 = (const float*)d_in[7];
    const float* ad1 = (const float*)d_in[8];
    const float* b1  = (const float*)d_in[9];

    float* out      = (float*)d_out;                       // [B,N,128]
    float* alphaOut = out + (size_t)BN * DM;               // [E]

    // workspace carve-out (256B aligned)
    char* base = (char*)d_ws;
    size_t off = 0;
    auto alloc = [&](size_t bytes) -> char* {
        char* p = base + off;
        off += (bytes + 255) & ~(size_t)255;
        return p;
    };
    _Float16* xf16  = (_Float16*)alloc((size_t)BN * DM * 2);          //  41 MB
    _Float16* w0t   = (_Float16*)alloc((size_t)DM * (NH * DM) * 2);   // 256 KB  [1024,128]
    _Float16* w1t   = (_Float16*)alloc((size_t)(NH * DM) * DM * 2);   // 256 KB  [128,1024]
    _Float16* h0f16 = (_Float16*)alloc((size_t)BN * NH * DM * 2);     // 328 MB (reused as x1)
    float*    e0s   = (float*)alloc((size_t)BN * NH * 4);
    float*    e0d   = (float*)alloc((size_t)BN * NH * 4);
    unsigned* m0u   = (unsigned*)alloc((size_t)BN * NH * 4);
    float*    d0    = (float*)alloc((size_t)BN * NH * 4);
    float*    ex0   = (float*)alloc((size_t)NB * NE * NH * 4);        // 164 MB
    float*    y0    = (float*)alloc((size_t)BN * NH * DM * 4);        // 655 MB
    float*    h1    = (float*)alloc((size_t)BN * DM * 4);             //  82 MB
    float*    e1s   = (float*)alloc((size_t)BN * 4);
    float*    e1d   = (float*)alloc((size_t)BN * 4);
    unsigned* m1u   = (unsigned*)alloc((size_t)BN * 4);
    float*    d1    = (float*)alloc((size_t)BN * 4);
    float*    ex1   = (float*)alloc((size_t)NB * NE * 4);             //  20 MB

    auto g = [](long long n, int b) { return (unsigned)((n + b - 1) / b); };

    // ---- precision prep ------------------------------------------------
    k_f32_to_f16<<<g((long long)BN * DM, 256), 256, 0, stream>>>(x, xf16, BN * DM);
    k_transpose_f16<<<g(DM * NH * DM, 256), 256, 0, stream>>>(W0, w0t, DM, NH * DM);
    k_transpose_f16<<<g(NH * DM * DM, 256), 256, 0, stream>>>(W1, w1t, NH * DM, DM);

    // ---- layer 0 -------------------------------------------------------
    // h0 = x @ W0 -> f16 [BN, 1024]
    k_gemm_wmma_f16<<<dim3(BN / 16, (NH * DM) / 256), 128, 0, stream>>>(
        xf16, w0t, h0f16, nullptr, DM, NH * DM);
    // attention dots per (row, head)
    k_edot<_Float16><<<g((long long)BN * NH * 32, 256), 256, 0, stream>>>(
        h0f16, as0, ad0, e0s, e0d, BN * NH, NH);
    // segment softmax
    k_fill_u32<<<g(BN * NH, 256), 256, 0, stream>>>(m0u, ENC_NEG_INF, BN * NH);
    hipMemsetAsync(d0, 0, (size_t)BN * NH * 4, stream);
    k_edge_max<<<g((long long)NB * NE * NH, 256), 256, 0, stream>>>(
        src, dst, e0s, e0d, m0u, NH, NB * NE * NH);
    k_edge_exp<<<g((long long)NB * NE * NH, 256), 256, 0, stream>>>(
        src, dst, e0s, e0d, m0u, ex0, d0, NH, NB * NE * NH);
    // aggregate: y0 = b0 + sum alpha*h[src]; also alpha_norm accumulation
    k_bias_bcast<<<g((long long)BN * NH * DM, 256), 256, 0, stream>>>(
        y0, b0, NH * DM, BN * NH * DM);
    hipMemsetAsync(alphaOut, 0, (size_t)NE * 4, stream);
    k_agg<true><<<g((long long)NB * NE * NH * 32, 256), 256, 0, stream>>>(
        src, dst, ex0, d0, h0f16, y0, alphaOut, 1.0f / (NB * NH),
        NH, NB * NE * NH * 32);
    // x1 = elu(y0) as f16 (overwrites h0 buffer, no longer needed)
    k_elu_f16<<<g((long long)BN * NH * DM, 256), 256, 0, stream>>>(
        y0, h0f16, BN * NH * DM);

    // ---- layer 1 -------------------------------------------------------
    // h1 = x1 @ W1 -> f32 [BN, 128]
    k_gemm_wmma_f16<<<dim3(BN / 16, 1), 64, 0, stream>>>(
        h0f16, w1t, nullptr, h1, NH * DM, DM);
    k_edot<float><<<g((long long)BN * 32, 256), 256, 0, stream>>>(
        h1, as1, ad1, e1s, e1d, BN, 1);
    k_fill_u32<<<g(BN, 256), 256, 0, stream>>>(m1u, ENC_NEG_INF, BN);
    hipMemsetAsync(d1, 0, (size_t)BN * 4, stream);
    k_edge_max<<<g((long long)NB * NE, 256), 256, 0, stream>>>(
        src, dst, e1s, e1d, m1u, 1, NB * NE);
    k_edge_exp<<<g((long long)NB * NE, 256), 256, 0, stream>>>(
        src, dst, e1s, e1d, m1u, ex1, d1, 1, NB * NE);
    k_bias_bcast<<<g((long long)BN * DM, 256), 256, 0, stream>>>(
        out, b1, DM, BN * DM);
    k_agg<false><<<g((long long)NB * NE * 32, 256), 256, 0, stream>>>(
        src, dst, ex1, d1, h1, out, nullptr, 0.f, 1, NB * NE * 32);
    (void)in_sizes; (void)n_in; (void)out_size; (void)ws_size;
}
